// causal_attention_57389353009366
// MI455X (gfx1250) — matrix operand order repaired and checked by reading, not compiled, not run on hardware
//
#include <hip/hip_runtime.h>
#include <hip/hip_bf16.h>

// ---------------------------------------------------------------------------
// Causal attention (strict mask), B=8, C=128, N=4096, fp32 in/out.
// Roofline: ~69 GFLOP vs ~67MB traffic -> compute bound -> use f16 WMMA
// (16x16x32, f32 accum) instead of f32 WMMA (16x16x4 = 8x more instructions).
// Pass 1: transpose [B,C,N] -> [B,N,C] + cvt f32->f16 into d_ws (24 MB).
// Pass 2: fused flash attention, K tiles staged by the Tensor Data Mover
//         (tensor_load_to_lds with LDS row padding), V staged transposed.
// ---------------------------------------------------------------------------

typedef __attribute__((ext_vector_type(16))) _Float16     v16h;
typedef __attribute__((ext_vector_type(8)))  float        v8f;
typedef __attribute__((ext_vector_type(4)))  unsigned int u32x4;
typedef __attribute__((ext_vector_type(8)))  int          i32x8;
typedef __attribute__((ext_vector_type(4)))  int          i32x4;

#define N_SEQ 4096
#define DHEAD 128
#define BM    128      // query rows per workgroup (8 waves x 16)
#define BN    64       // key/value columns per tile
#define NWAVE 8

// LDS strides in halves: 16B-aligned rows, stride(dwords) % 64 chosen for
// conflict-free b128 access across the 16-lane groups.
#define KSTR  136      // 128 + 4dw pad  (68 dw = 4 mod 64)
#define VSTR  72       // 64  + 4dw pad  (36 dw = 36 mod 64, gcd(36,64)=4 -> ok)
#define PSTR  72

#define K_TILE  (BN * KSTR)              // 8704 halves / buffer
#define VT_TILE (DHEAD * VSTR)           // 9216 halves / buffer
#define P_TILE  (16 * PSTR)              // 1152 halves / wave
#define K_OFF   0
#define VT_OFF  (2 * K_TILE)             // 17408
#define P_OFF   (VT_OFF + 2 * VT_TILE)   // 35840
#define SMEM_HALVES (P_OFF + NWAVE * P_TILE)  // 45056 halves = 88 KB

// ---------------------------------------------------------------------------
// Pass 1: [B, C, N] f32  ->  [B, N, C] f16   (64x64 LDS tile transpose)
// ---------------------------------------------------------------------------
__global__ __launch_bounds__(256)
void transpose_cvt_kernel(const float* __restrict__ src, _Float16* __restrict__ dst) {
    __shared__ _Float16 tile[64 * 65];
    const int b  = blockIdx.z;
    const int c0 = blockIdx.y * 64;
    const int n0 = blockIdx.x * 64;
    const float*    sb = src + ((size_t)b * DHEAD + c0) * N_SEQ + n0;
    _Float16*       db = dst + ((size_t)b * N_SEQ + n0) * DHEAD + c0;
#pragma unroll
    for (int p = 0; p < 16; ++p) {
        int l = p * 256 + threadIdx.x;
        int i = l >> 6, j = l & 63;                    // coalesced along n
        tile[i * 65 + j] = (_Float16)sb[(size_t)i * N_SEQ + j];
    }
    __syncthreads();
#pragma unroll
    for (int p = 0; p < 16; ++p) {
        int l = p * 256 + threadIdx.x;
        int i = l & 63, j = l >> 6;                    // coalesced along c
        db[(size_t)j * DHEAD + i] = tile[i * 65 + j];
    }
}

// ---------------------------------------------------------------------------
// TDM: stage one 64x128 f16 K tile into LDS, padding each 128-half row to
// KSTR=136 halves (pad_interval: 64 dwords -> code 5, pad_amount: 4 dw -> 3).
// D# per CDNA5 ISA. Issued by wave 0 only.
// This toolchain's builtin takes 6 args:
//   (uint32x4 g0, int32x8 g1, int32x4 g2, int32x4 g3, int32x8 extra, i32 cpol)
// ---------------------------------------------------------------------------
__device__ __forceinline__ void tdm_load_k_tile(const _Float16* gsrc, unsigned lds_byte_off) {
    unsigned long long ga = (unsigned long long)(const void*)gsrc;
    u32x4 g0;
    g0[0] = 1u;                                         // count=1, user mode
    g0[1] = lds_byte_off;                               // lds_addr
    g0[2] = (unsigned)ga;                               // global_addr[31:0]
    g0[3] = (unsigned)((ga >> 32) & 0x1FFFFFFu) | (2u << 30);  // [56:32] | type=2
    i32x8 g1;
    g1[0] = (int)((1u << 16) |                          // data_size = 2 bytes
                  (1u << 20) |                          // pad_enable
                  (5u << 22) |                          // pad_interval: 64 dw
                  (3u << 25));                          // pad_amount: 4 dw
    g1[1] = (int)(128u << 16);                          // tensor_dim0 = 128
    g1[2] = (int)((unsigned)N_SEQ << 16);               // tensor_dim1 lo
    g1[3] = (int)(128u << 16);                          // tile_dim0 = 128
    g1[4] = 64;                                         // tile_dim1=64, tile_dim2=0
    g1[5] = 128;                                        // tensor_dim0_stride = 128
    g1[6] = 0;
    g1[7] = 0;
    i32x4 z4; z4[0] = 0; z4[1] = 0; z4[2] = 0; z4[3] = 0;   // 2D: groups 2/3 unused
    i32x8 z8;
#pragma unroll
    for (int i = 0; i < 8; ++i) z8[i] = 0;
    __builtin_amdgcn_tensor_load_to_lds(g0, g1, z4, z4, z8, 0);
}

// Stage a 64x128 f16 V tile transposed into LDS: Vt[c][s], row stride VSTR.
__device__ __forceinline__ void stage_v_tile(const _Float16* __restrict__ vsrc,
                                             _Float16* __restrict__ smem,
                                             int bi, int tid) {
    _Float16* vt = smem + VT_OFF + bi * VT_TILE;
#pragma unroll
    for (int p = 0; p < 4; ++p) {
        int g  = tid + p * 256;        // 0..1023 groups of 8 halves
        int s  = g >> 4;               // seq row   0..63
        int c0 = (g & 15) * 8;         // d column
        union { uint4 u; _Float16 h[8]; } t;
        t.u = *(const uint4*)(vsrc + (size_t)s * DHEAD + c0);
#pragma unroll
        for (int e = 0; e < 8; ++e)
            vt[(c0 + e) * VSTR + s] = t.h[e];
    }
}

// ---------------------------------------------------------------------------
// Pass 2: fused causal flash attention.  grid = (N/BM, B), block = 256.
// ---------------------------------------------------------------------------
__global__ __launch_bounds__(256)
void flash_causal_kernel(const _Float16* __restrict__ qh,
                         const _Float16* __restrict__ kh,
                         const _Float16* __restrict__ mh,
                         float* __restrict__ out) {
    __shared__ __attribute__((aligned(32))) _Float16 smem[SMEM_HALVES];

    const int tid  = threadIdx.x;
    const int wave = tid >> 5;
    const int lane = tid & 31;
    const int l15  = lane & 15;
    const int hi   = lane >> 4;            // lane-half select (VGPR layout)
    const int b    = blockIdx.y;
    const int i0   = blockIdx.x * BM;
    const int jt_max = (i0 >> 6) + 2;      // strict causal: tiles with j <= i0+126

    const _Float16* qb = qh + (size_t)b * N_SEQ * DHEAD;
    const _Float16* kb = kh + (size_t)b * N_SEQ * DHEAD;
    const _Float16* mb = mh + (size_t)b * N_SEQ * DHEAD;

    // ---- Q A-fragments (16x32 chunks, ISA 16-bit A layout: lo lanes K{0..7,16..23})
    v16h qfrag[4];
    {
        const _Float16* qrow = qb + (size_t)(i0 + wave * 16 + l15) * DHEAD;
#pragma unroll
        for (int kc = 0; kc < 4; ++kc) {
            union { v16h h; uint4 u[2]; } t;
            const _Float16* p = qrow + kc * 32 + hi * 8;
            t.u[0] = *(const uint4*)(p);
            t.u[1] = *(const uint4*)(p + 16);
            qfrag[kc] = t.h;
        }
    }

    v8f o_acc[8];
#pragma unroll
    for (int ng = 0; ng < 8; ++ng) {
        v8f z = {0.f, 0.f, 0.f, 0.f, 0.f, 0.f, 0.f, 0.f};
        o_acc[ng] = z;
    }
    float rmax[8], rsum[8];
#pragma unroll
    for (int r = 0; r < 8; ++r) { rmax[r] = -3.0e38f; rsum[r] = 0.f; }

    _Float16* pbuf = smem + P_OFF + wave * P_TILE;
    const float inv_s = 0.08838834764831845f;   // 1/sqrt(128)

    // Prologue: stage tile 0 (K via TDM, V transposed by all threads).
    if (wave == 0) tdm_load_k_tile(kb, (unsigned)(K_OFF * 2));
    stage_v_tile(mb, smem, 0, tid);

    for (int jt = 0; jt < jt_max; ++jt) {
        const int bi = jt & 1;
        const int j0 = jt * BN;

        if (wave == 0) __builtin_amdgcn_s_wait_tensorcnt(0);
        __syncthreads();   // tile jt resident in LDS for everyone

        // Pipeline: kick off staging of tile jt+1 into the other buffer.
        if (jt + 1 < jt_max) {
            const size_t goff = (size_t)(jt + 1) * BN * DHEAD;
            if (wave == 0)
                tdm_load_k_tile(kb + goff, (unsigned)((K_OFF + (bi ^ 1) * K_TILE) * 2));
            stage_v_tile(mb + goff, smem, bi ^ 1, tid);
        }

        // ---- S = Q * K^T   (4 n-groups x 4 k-chunks of WMMA f16->f32)
        const _Float16* kt = smem + K_OFF + bi * K_TILE;
        v8f s_acc[4];
#pragma unroll
        for (int nt = 0; nt < 4; ++nt) {
            v8f c = {0.f, 0.f, 0.f, 0.f, 0.f, 0.f, 0.f, 0.f};
#pragma unroll
            for (int kc = 0; kc < 4; ++kc) {
                union { v16h h; uint4 u[2]; } t;   // B frag: lane=col, K{8h..8h+7, 16+8h..16+8h+7}
                const _Float16* p = kt + (nt * 16 + l15) * KSTR + kc * 32 + hi * 8;
                t.u[0] = *(const uint4*)p;
                t.u[1] = *(const uint4*)(p + 16);
                c = __builtin_amdgcn_wmma_f32_16x16x32_f16(
                        false, qfrag[kc], false, t.h, (short)0, c, false, false);
            }
            s_acc[nt] = c;
        }

        // ---- online softmax, matching reference semantics exactly:
        // mask as s-1e10 BEFORE row max, exp((s-m)/sqrt(d)), p=0 on mask.
        const bool diag = (j0 >= i0);
#pragma unroll
        for (int r = 0; r < 8; ++r) {
            const int irow = i0 + wave * 16 + r + hi * 8;
            float sv[4];
#pragma unroll
            for (int nt = 0; nt < 4; ++nt) {
                float s = s_acc[nt][r];
                if (diag && (j0 + nt * 16 + l15) >= irow) s -= 1.0e10f;
                sv[nt] = s;
            }
            float tmax = fmaxf(fmaxf(sv[0], sv[1]), fmaxf(sv[2], sv[3]));
#pragma unroll
            for (int off = 8; off >= 1; off >>= 1)
                tmax = fmaxf(tmax, __shfl_xor(tmax, off, 32));
            const float mnew = fmaxf(rmax[r], tmax);
            const float corr = __expf((rmax[r] - mnew) * inv_s);
            rmax[r] = mnew;

            float psum = 0.f;
#pragma unroll
            for (int nt = 0; nt < 4; ++nt) {
                float p = __expf((sv[nt] - mnew) * inv_s);
                if (diag && (j0 + nt * 16 + l15) >= irow) p = 0.f;  // * mask
                psum += p;
                pbuf[(r + hi * 8) * PSTR + nt * 16 + l15] = (_Float16)p;
            }
#pragma unroll
            for (int off = 8; off >= 1; off >>= 1)
                psum += __shfl_xor(psum, off, 32);
            rsum[r] = rsum[r] * corr + psum;
#pragma unroll
            for (int ng = 0; ng < 8; ++ng) o_acc[ng][r] *= corr;
        }

        // ---- O += P * V   (re-tile P via per-wave LDS scratch; V from Vt)
        const _Float16* vt = smem + VT_OFF + bi * VT_TILE;
#pragma unroll
        for (int kc = 0; kc < 2; ++kc) {
            union { v16h h; uint4 u[2]; } pa;      // A frag of P (16x32 chunk)
            const _Float16* pp = pbuf + l15 * PSTR + kc * 32 + hi * 8;
            pa.u[0] = *(const uint4*)pp;
            pa.u[1] = *(const uint4*)(pp + 16);
#pragma unroll
            for (int ng = 0; ng < 8; ++ng) {
                union { v16h h; uint4 u[2]; } vb;  // B frag from transposed V
                const _Float16* vp = vt + (ng * 16 + l15) * VSTR + kc * 32 + hi * 8;
                vb.u[0] = *(const uint4*)vp;
                vb.u[1] = *(const uint4*)(vp + 16);
                o_acc[ng] = __builtin_amdgcn_wmma_f32_16x16x32_f16(
                        false, pa.h, false, vb.h, (short)0, o_acc[ng], false, false);
            }
        }
        __syncthreads();   // all reads of buffer bi done before it is re-staged
    }

    // ---- epilogue: out[b][c][i] = O / (l + 1e-6)   (output is [B,C,N])
    float* ob = out + (size_t)b * DHEAD * N_SEQ;
#pragma unroll
    for (int ng = 0; ng < 8; ++ng) {
        const int c = ng * 16 + l15;
#pragma unroll
        for (int r = 0; r < 8; ++r) {
            const int irow = i0 + wave * 16 + r + hi * 8;
            ob[(size_t)c * N_SEQ + irow] = o_acc[ng][r] / (rsum[r] + 1e-6f);
        }
    }
}

// ---------------------------------------------------------------------------
extern "C" void kernel_launch(void* const* d_in, const int* in_sizes, int n_in,
                              void* d_out, int out_size, void* d_ws, size_t ws_size,
                              hipStream_t stream) {
    (void)in_sizes; (void)n_in; (void)out_size; (void)ws_size;
    const float* key   = (const float*)d_in[0];
    const float* mixin = (const float*)d_in[1];
    const float* query = (const float*)d_in[2];

    const size_t per = (size_t)8 * DHEAD * N_SEQ;     // 4,194,304 elements
    _Float16* qh = (_Float16*)d_ws;                   // needs 3*per*2 = 24 MB
    _Float16* kh = qh + per;
    _Float16* mh = kh + per;

    dim3 tgrid(N_SEQ / 64, DHEAD / 64, 8);
    transpose_cvt_kernel<<<tgrid, 256, 0, stream>>>(query, qh);
    transpose_cvt_kernel<<<tgrid, 256, 0, stream>>>(key,   kh);
    transpose_cvt_kernel<<<tgrid, 256, 0, stream>>>(mixin, mh);

    dim3 fgrid(N_SEQ / BM, 8);
    flash_causal_kernel<<<fgrid, 256, 0, stream>>>(qh, kh, mh, (float*)d_out);
}
